// MulitiHeadAttention_81226421502612
// MI455X (gfx1250) — matrix operand order repaired and checked by reading, not compiled, not run on hardware
//
#include <hip/hip_runtime.h>
#include <hip/hip_bf16.h>
#include <math.h>
#include <stdint.h>

// ---------------------------------------------------------------------------
// MI455X / gfx1250 fused conv1x1-projection + flash-attention + out-projection
// Matrix math: V_WMMA_F32_16X16X32_BF16 (wave32). K/V tiles staged into LDS
// via the Tensor Data Mover (TENSOR_LOAD_TO_LDS + s_wait_tensorcnt), double
// buffered so the DMA of chunk j+1 overlaps softmax+WMMA of chunk j.
// ---------------------------------------------------------------------------

typedef __bf16 bf16_t;
typedef __attribute__((ext_vector_type(16))) __bf16 v16bf;
typedef __attribute__((ext_vector_type(8)))  float  v8f;

#define NPOS   4096   // H*W
#define CCH    256    // channels
#define NHEAD  8
#define DHEAD  32
#define BATCH  2
#define KCHUNK 32                      // keys per attention inner step
#define CHUNK_ELEMS (KCHUNK * DHEAD)   // 1024 bf16 = 2 KB per K/V chunk

// ---------------------------------------------------------------------------
// Tensor Data Mover: contiguous 2KB tile copy global -> LDS.
//   D# group0: count=1 | lds_addr | global_addr | type=2
//   D# group1: data_size=1 (2B), tensor_dim0 = tile_dim0 = 1024 elements
// ---------------------------------------------------------------------------
#if __has_builtin(__builtin_amdgcn_tensor_load_to_lds)
#define USE_TDM 1
typedef uint32_t u32x4_t __attribute__((ext_vector_type(4)));
typedef int      i32x4_t __attribute__((ext_vector_type(4)));
typedef int      i32x8_t __attribute__((ext_vector_type(8)));

__device__ __forceinline__ void tdm_copy_chunk_to_lds(const bf16_t* gsrc,
                                                      uint32_t lds_byte_off) {
    const uint64_t ga = (uint64_t)(uintptr_t)gsrc;
    u32x4_t g0;
    g0[0] = 1u;                                   // count=1, user descriptor
    g0[1] = lds_byte_off;                         // lds_addr[31:0]
    g0[2] = (uint32_t)ga;                         // global_addr[31:0]
    g0[3] = (uint32_t)(ga >> 32) | (2u << 30);    // global_addr[56:32] | type=2
    i32x8_t g1;
    g1[0] = (1 << 16);                            // data_size=1 (2B), mask=0
    g1[1] = (int)((CHUNK_ELEMS & 0xFFFF) << 16);  // tensor_dim0[15:0]
    g1[2] = (int)((CHUNK_ELEMS >> 16) & 0xFFFF)   // tensor_dim0[31:16]
          | (1 << 16);                            // tensor_dim1 = 1
    g1[3] = (int)(CHUNK_ELEMS << 16);             // tile_dim0 = 1024
    g1[4] = 0;                                    // tile_dim1/2 unused
    g1[5] = (int)CHUNK_ELEMS;                     // tensor_dim0_stride[31:0]
    g1[6] = 0;
    g1[7] = 0;
    const i32x4_t gz = {0, 0, 0, 0};
#if defined(__clang_major__) && (__clang_major__ >= 23)
    const i32x8_t gz8 = {0, 0, 0, 0, 0, 0, 0, 0}; // therock-10.0 6-arg form
    __builtin_amdgcn_tensor_load_to_lds(g0, g1, gz, gz, gz8, 0);
#else
    __builtin_amdgcn_tensor_load_to_lds(g0, g1, gz, gz, 0);  // ROCm 7.2 5-arg
#endif
}
#else
#define USE_TDM 0
#endif

// ---------------------------------------------------------------------------
// Projection GEMM:  Y[bh, n, d] = bf16( sum_c W[o,c] * X[b,c,n] + bias[o] )
// One wave computes a 16(o) x 16(n) tile, K=256 in 8 WMMAs.
// A-fragment (16x32 bf16): lane m = lane&15, g = lane>>4, e -> k=e+16g
// B-fragment (32x16 bf16): lane n = lane&15, g = lane>>4, e -> k=e+16g
// C/D (16x16 f32): lane n = lane&15, VGPR r -> row = r + 8*(lane>>4)
// ---------------------------------------------------------------------------
__global__ __launch_bounds__(128)
void proj_qkv_kernel(const float* __restrict__ X,    // (B, 256, 4096)
                     const float* __restrict__ W,    // (256, 256)
                     const float* __restrict__ bias, // (256,)
                     bf16_t* __restrict__ Y)         // (BH, 4096, 32) bf16
{
    __shared__ bf16_t ldsW[16 * CCH];                // 8 KB: 16 o-rows x 256 k
    const int lane = threadIdx.x & 31;
    const int wave = threadIdx.x >> 5;
    const int o0   = blockIdx.y * 16;
    const int b    = blockIdx.z;

    for (int i = threadIdx.x; i < 16 * CCH; i += 128) {
        const int r = i >> 8;
        const int k = i & 255;
        ldsW[i] = (bf16_t)W[(o0 + r) * CCH + k];
    }
    __syncthreads();

    const int n0 = blockIdx.x * 64 + wave * 16;
    const int nl = lane & 15;
    const int g  = lane >> 4;
    const float* __restrict__ Xb = X + (size_t)b * CCH * NPOS;

    v8f acc = {0.f, 0.f, 0.f, 0.f, 0.f, 0.f, 0.f, 0.f};
    #pragma unroll
    for (int k0 = 0; k0 < CCH; k0 += 32) {
        v16bf a, bb;
        #pragma unroll
        for (int e = 0; e < 16; ++e) {
            const int kk = e + 16 * g;
            a[e] = ldsW[nl * CCH + k0 + kk];
        }
        #pragma unroll
        for (int e = 0; e < 16; ++e) {
            const int k = k0 + e + 16 * g;
            bb[e] = (bf16_t)Xb[(size_t)k * NPOS + (n0 + nl)];
        }
        acc = __builtin_amdgcn_wmma_f32_16x16x32_bf16(false, a, false, bb,
                                                      (short)0, acc, false, false);
    }

    #pragma unroll
    for (int r = 0; r < 8; ++r) {
        const int o = o0 + r + 8 * g;
        const float v = acc[r] + bias[o];
        const int head = o >> 5, d = o & 31;
        const int bh = b * NHEAD + head;
        Y[((size_t)bh * NPOS + (n0 + nl)) * DHEAD + d] = (bf16_t)v;
    }
}

// ---------------------------------------------------------------------------
// Flash attention. One wave per 16-query tile; 4 waves/block share the same
// (bh, key-chunk) stream, so K/V chunks are staged once per block in LDS by
// the TDM, double buffered.  S = Q Kt (unscaled), online softmax, O = P V,
// final scale (1/sqrt(32))/l applied AFTER softmax per the reference quirk.
// ---------------------------------------------------------------------------
__global__ __launch_bounds__(128)
void flash_attn_kernel(const bf16_t* __restrict__ Q,   // (BH, N, 32)
                       const bf16_t* __restrict__ K,
                       const bf16_t* __restrict__ V,
                       bf16_t* __restrict__ O)          // (BH, N, 32)
{
    __shared__ __align__(16) bf16_t ldsK[2][CHUNK_ELEMS];  // 2 x 2 KB
    __shared__ __align__(16) bf16_t ldsV[2][CHUNK_ELEMS];  // 2 x 2 KB
    __shared__ bf16_t ldsP[4][16 * 32];                    // per-wave P tile

    const int lane = threadIdx.x & 31;
    const int wave = threadIdx.x >> 5;
    const int bh   = blockIdx.y;
    const int q0   = (blockIdx.x * 4 + wave) * 16;
    const int nl   = lane & 15;
    const int g    = lane >> 4;

    const bf16_t* __restrict__ Qh = Q + (size_t)bh * NPOS * DHEAD;
    const bf16_t* __restrict__ Kh = K + (size_t)bh * NPOS * DHEAD;
    const bf16_t* __restrict__ Vh = V + (size_t)bh * NPOS * DHEAD;

    // stage one 32-key K/V chunk into LDS buffer `buf`
    auto stage = [&](int buf, int j0) {
#if USE_TDM
        if (wave == 0) {   // one TDM descriptor pair per workgroup
            tdm_copy_chunk_to_lds(Kh + (size_t)j0 * DHEAD,
                                  (uint32_t)(uintptr_t)&ldsK[buf][0]);
            tdm_copy_chunk_to_lds(Vh + (size_t)j0 * DHEAD,
                                  (uint32_t)(uintptr_t)&ldsV[buf][0]);
        }
#else
        // fallback: cooperative 16B/thread copy (128 threads x 16B = 2 KB)
        const uint4* __restrict__ gk = (const uint4*)(Kh + (size_t)j0 * DHEAD);
        const uint4* __restrict__ gv = (const uint4*)(Vh + (size_t)j0 * DHEAD);
        ((uint4*)&ldsK[buf][0])[threadIdx.x] = gk[threadIdx.x];
        ((uint4*)&ldsV[buf][0])[threadIdx.x] = gv[threadIdx.x];
#endif
    };

    // Q A-fragment, loaded once (one contiguous 32B bf16 run per lane).
    v16bf qa;
    #pragma unroll
    for (int e = 0; e < 16; ++e) {
        const int d = e + 16 * g;
        qa[e] = Qh[(size_t)(q0 + nl) * DHEAD + d];
    }

    float mrow[8], lrow[8];
    #pragma unroll
    for (int r = 0; r < 8; ++r) { mrow[r] = -3.0e38f; lrow[r] = 0.f; }
    v8f oa0 = {0.f,0.f,0.f,0.f,0.f,0.f,0.f,0.f};
    v8f oa1 = {0.f,0.f,0.f,0.f,0.f,0.f,0.f,0.f};
    const v8f zero8 = {0.f,0.f,0.f,0.f,0.f,0.f,0.f,0.f};

    stage(0, 0);
    for (int jc = 0; jc < NPOS / KCHUNK; ++jc) {
        const int cur = jc & 1;
        // issue DMA for next chunk, then wait only for the current one
        if (jc + 1 < NPOS / KCHUNK) {
            stage(cur ^ 1, (jc + 1) * KCHUNK);
#if USE_TDM
            if (wave == 0) __builtin_amdgcn_s_wait_tensorcnt(2);
#endif
        } else {
#if USE_TDM
            if (wave == 0) __builtin_amdgcn_s_wait_tensorcnt(0);
#endif
        }
        __syncthreads();               // current buffer visible to all waves

        // ---- S = Q * K^T : two 16-key column tiles (K rows: 32B ds reads) ----
        v8f s[2];
        #pragma unroll
        for (int jt = 0; jt < 2; ++jt) {
            v16bf kb;
            const bf16_t* __restrict__ kr = &ldsK[cur][(jt * 16 + nl) * DHEAD + 16 * g];
            #pragma unroll
            for (int e = 0; e < 16; ++e) kb[e] = kr[e];
            s[jt] = __builtin_amdgcn_wmma_f32_16x16x32_bf16(false, qa, false, kb,
                                                            (short)0, zero8, false, false);
        }

        // ---- online softmax: row max over 16 lanes of each half-group ----
        float mnew[8], alpha[8], tsum[8];
        #pragma unroll
        for (int r = 0; r < 8; ++r) {
            float mx = fmaxf(s[0][r], s[1][r]);
            #pragma unroll
            for (int off = 8; off >= 1; off >>= 1)
                mx = fmaxf(mx, __shfl_xor(mx, off, 32));
            mnew[r]  = fmaxf(mrow[r], mx);
            alpha[r] = __expf(mrow[r] - mnew[r]);
        }
        #pragma unroll
        for (int r = 0; r < 8; ++r) {
            const float p0 = __expf(s[0][r] - mnew[r]);
            const float p1 = __expf(s[1][r] - mnew[r]);
            tsum[r] = p0 + p1;
            const int row = r + 8 * g;
            ldsP[wave][row * 32 + nl]      = (bf16_t)p0;
            ldsP[wave][row * 32 + 16 + nl] = (bf16_t)p1;
        }
        #pragma unroll
        for (int r = 0; r < 8; ++r) {
            float ts = tsum[r];
            #pragma unroll
            for (int off = 8; off >= 1; off >>= 1)
                ts += __shfl_xor(ts, off, 32);
            lrow[r] = lrow[r] * alpha[r] + ts;
            mrow[r] = mnew[r];
            oa0[r] *= alpha[r];
            oa1[r] *= alpha[r];
        }

        // ---- P as A-fragment from LDS (same-wave DS ops are in-order) ----
        v16bf pa;
        #pragma unroll
        for (int e = 0; e < 16; ++e) {
            const int kk = e + 16 * g;
            pa[e] = ldsP[wave][nl * 32 + kk];
        }

        // ---- O += P * V : two 16-wide dv tiles, V from LDS ----
        #pragma unroll
        for (int dt = 0; dt < 2; ++dt) {
            v16bf vb;
            const int dv = dt * 16 + nl;
            #pragma unroll
            for (int e = 0; e < 16; ++e)
                vb[e] = ldsV[cur][(e + 16 * g) * DHEAD + dv];
            if (dt == 0)
                oa0 = __builtin_amdgcn_wmma_f32_16x16x32_bf16(false, pa, false, vb,
                                                              (short)0, oa0, false, false);
            else
                oa1 = __builtin_amdgcn_wmma_f32_16x16x32_bf16(false, pa, false, vb,
                                                              (short)0, oa1, false, false);
        }

        __syncthreads();               // all waves done with `cur` before reuse
    }

    const float sc = 0.17677669529663687f;            // 1/sqrt(32), post-softmax
    #pragma unroll
    for (int r = 0; r < 8; ++r) {
        const float inv = sc / lrow[r];
        const size_t row = (size_t)bh * NPOS + (q0 + r + 8 * g);
        O[row * DHEAD + nl]      = (bf16_t)(oa0[r] * inv);
        O[row * DHEAD + 16 + nl] = (bf16_t)(oa1[r] * inv);
    }
}

// ---------------------------------------------------------------------------
// Output projection: out[b,o,n] = sum_c W_out[o,c] * Oc[b, c, n] + b_out[o]
// ---------------------------------------------------------------------------
__global__ __launch_bounds__(128)
void out_proj_kernel(const bf16_t* __restrict__ Oc,  // (BH, 4096, 32)
                     const float* __restrict__ W,    // (256, 256)
                     const float* __restrict__ bias, // (256,)
                     float* __restrict__ out)        // (B, 256, 4096)
{
    __shared__ bf16_t ldsW[16 * CCH];
    const int lane = threadIdx.x & 31;
    const int wave = threadIdx.x >> 5;
    const int o0   = blockIdx.y * 16;
    const int b    = blockIdx.z;

    for (int i = threadIdx.x; i < 16 * CCH; i += 128) {
        const int r = i >> 8;
        const int k = i & 255;
        ldsW[i] = (bf16_t)W[(o0 + r) * CCH + k];
    }
    __syncthreads();

    const int n0 = blockIdx.x * 64 + wave * 16;
    const int nl = lane & 15;
    const int g  = lane >> 4;

    v8f acc = {0.f, 0.f, 0.f, 0.f, 0.f, 0.f, 0.f, 0.f};
    #pragma unroll
    for (int k0 = 0; k0 < CCH; k0 += 32) {           // one head per K-chunk
        v16bf a, bb;
        #pragma unroll
        for (int e = 0; e < 16; ++e) {
            const int kk = e + 16 * g;
            a[e] = ldsW[nl * CCH + k0 + kk];
        }
        const int head = k0 >> 5;
        const bf16_t* __restrict__ src =
            Oc + ((size_t)(b * NHEAD + head) * NPOS + (n0 + nl)) * DHEAD + 16 * g;
        #pragma unroll
        for (int e = 0; e < 16; ++e) bb[e] = src[e]; // contiguous 32B per lane
        acc = __builtin_amdgcn_wmma_f32_16x16x32_bf16(false, a, false, bb,
                                                      (short)0, acc, false, false);
    }

    #pragma unroll
    for (int r = 0; r < 8; ++r) {
        const int o = o0 + r + 8 * g;
        out[((size_t)b * CCH + o) * NPOS + (n0 + nl)] = acc[r] + bias[o];
    }
}

// ---------------------------------------------------------------------------
extern "C" void kernel_launch(void* const* d_in, const int* in_sizes, int n_in,
                              void* d_out, int out_size, void* d_ws, size_t ws_size,
                              hipStream_t stream) {
    const float* q_in  = (const float*)d_in[0];
    const float* k_in  = (const float*)d_in[1];
    const float* v_in  = (const float*)d_in[2];
    const float* w_q   = (const float*)d_in[3];
    const float* b_q   = (const float*)d_in[4];
    const float* w_k   = (const float*)d_in[5];
    const float* b_k   = (const float*)d_in[6];
    const float* w_v   = (const float*)d_in[7];
    const float* b_v   = (const float*)d_in[8];
    const float* w_out = (const float*)d_in[9];
    const float* b_out = (const float*)d_in[10];
    float* out = (float*)d_out;

    const size_t elems = (size_t)BATCH * NHEAD * NPOS * DHEAD;  // 2,097,152
    bf16_t* Qb = (bf16_t*)d_ws;          // 4 MB each, 16 MB total
    bf16_t* Kb = Qb + elems;
    bf16_t* Vb = Kb + elems;
    bf16_t* Ob = Vb + elems;

    const dim3 gp(64, 16, 2), bp(128);   // 4 waves/block, 1 wave = 16x16 tile
    proj_qkv_kernel<<<gp, bp, 0, stream>>>(q_in, w_q, b_q, Qb);
    proj_qkv_kernel<<<gp, bp, 0, stream>>>(k_in, w_k, b_k, Kb);
    proj_qkv_kernel<<<gp, bp, 0, stream>>>(v_in, w_v, b_v, Vb);

    flash_attn_kernel<<<dim3(64, 16), bp, 0, stream>>>(Qb, Kb, Vb, Ob);

    out_proj_kernel<<<gp, bp, 0, stream>>>(Ob, w_out, b_out, out);
}
